// SNN_87978110091907
// MI455X (gfx1250) — matrix operand order repaired and checked by reading, not compiled, not run on hardware
//
#include <hip/hip_runtime.h>

// ---------------------------------------------------------------------------
// SNN forward on MI455X (gfx1250), single persistent workgroup (one WGP).
//
// Shapes: V=50257, E=H=256, B=32, S=256, T=10.  S*T=2560 sequential steps.
// 16 waves (512 threads): wave = one 16-wide H-column tile, computing both
// 16-row batch tiles.
//   * W2 B-fragments: bf16, PERMANENT in VGPRs (64 VGPRs/wave).
//   * W3: bf16 in LDS (preconverted once); fragment loads kept in the loop
//     via an opaque integer offset so the allocator cannot hoist-and-spill.
//   * W1: preconverted f32->bf16 into d_ws by a tiny pre-kernel, so the
//     per-word delta1 GEMM does single 32B global bf16 loads (L2-resident).
//   * spikes: double-buffered 16KB LDS matrices, 2 barriers/timestep.
// Matrix path: v_wmma_f32_16x16x32_bf16 (f32 accumulate).
// ---------------------------------------------------------------------------

typedef __attribute__((ext_vector_type(16))) __bf16 v16bf;
typedef __attribute__((ext_vector_type(8)))  __bf16 v8bf;
typedef __attribute__((ext_vector_type(8)))  float  v8f;
typedef __attribute__((ext_vector_type(16))) float  v16f;

#define HD 256   // hidden / embed dim
#define BB 32    // batch
#define SS 256   // sequence length
#define TT 10    // LIF timesteps per word

// Defeat LICM/CSE of loads addressed with this offset, WITHOUT destroying
// pointer address-space inference (keeps global_load / ds_load selection).
__device__ __forceinline__ int opaque_i(int v) {
    asm volatile("" : "+v"(v));
    return v;
}

// A fragment (16x32 bf16, row-major source), contiguous k order matching B:
// lanes 0-15 hold K=[0..16), lanes 16-31 hold K=[16..32).
__device__ __forceinline__ v16bf load_afrag(const __bf16* buf, int row, int k0, int half) {
    const __bf16* base = buf + row * HD + k0 + half * 16;
    v8bf lo = *(const v8bf*)(base);
    v8bf hi = *(const v8bf*)(base + 8);
    return __builtin_shufflevector(lo, hi, 0,1,2,3,4,5,6,7,8,9,10,11,12,13,14,15);
}

// B fragment (32x16 bf16) from LDS: lane holds W[n, k0+half*16 .. +16].
__device__ __forceinline__ v16bf load_bfrag_lds(const __bf16* w, int n, int k0, int half) {
    return *(const v16bf*)(w + n * HD + opaque_i(k0 + half * 16));
}

// B fragment from a row-major bf16 matrix in global (single 32B load).
__device__ __forceinline__ v16bf load_bfrag_glb_bf(const __bf16* w, int n, int k0, int half) {
    return *(const v16bf*)(w + n * HD + opaque_i(k0 + half * 16));
}

// B fragment from a row-major f32 matrix in global: one 64B vector load + cvt.
__device__ __forceinline__ v16bf load_bfrag_glb_f32(const float* w, int n, int k0, int half) {
    const float* p = w + n * HD + opaque_i(k0 + half * 16);
    v16f t = *(const v16f*)p;
    v16bf r;
#pragma unroll
    for (int j = 0; j < 16; ++j) r[j] = (__bf16)t[j];
    return r;
}

// ---- pre-kernel: W1 f32 -> bf16 into workspace (runs once per launch) ----
__global__ void W1_to_bf16_kernel(const float* __restrict__ src,
                                  __bf16* __restrict__ dst) {
    int i = blockIdx.x * blockDim.x + threadIdx.x;   // 65536 threads
    dst[i] = (__bf16)src[i];
}

__global__ __launch_bounds__(512, 1) void SNN_persistent_kernel(
    const int*   __restrict__ text,
    const float* __restrict__ emb,
    const __bf16* __restrict__ W1bf, const float* __restrict__ b1,
    const float* __restrict__ thr1p, const float* __restrict__ leak1p,
    const float* __restrict__ W2,  const float* __restrict__ b2,
    const float* __restrict__ thr2p, const float* __restrict__ leak2p,
    const float* __restrict__ W3,  const float* __restrict__ b3,
    const float* __restrict__ Wa1, const float* __restrict__ ba1,
    const float* __restrict__ Wa2, const float* __restrict__ ba2,
    float* __restrict__ out)
{
    extern __shared__ __align__(16) unsigned char smem[];
    __bf16* sX  = (__bf16*)(smem);            //  16 KB  x word [32,256] bf16
    __bf16* sS1 = (__bf16*)(smem + 16384);    //  16 KB  spikes s1
    __bf16* sS2 = (__bf16*)(smem + 32768);    //  16 KB  spikes s2
    __bf16* sW3 = (__bf16*)(smem + 49152);    // 128 KB  W3 bf16

    const int tid   = threadIdx.x;
    const int lane  = tid & 31;
    const int laneM = lane & 15;
    const int half  = lane >> 4;               // 0/1
    const int tc    = tid >> 5;                // wave = H-column tile 0..15
    const int col   = tc * 16 + laneM;         // output column owned by lane

    const unsigned short ONE_BF = 0x3F80;      // bf16(1.0)
    unsigned short* sS1u = (unsigned short*)sS1;
    unsigned short* sS2u = (unsigned short*)sS2;

    // ---- phase 0: W3 f32 -> bf16 into LDS (128 elems/thread) ----
#pragma unroll 4
    for (int k = 0; k < 128; ++k) { int i = (k << 9) + tid; sW3[i] = (__bf16)W3[i]; }

    // ---- W2 B-fragments: persistent in VGPRs (f32 -> bf16 once) ----
    v16bf Bw2[8];
#pragma unroll
    for (int kk = 0; kk < 8; ++kk) Bw2[kk] = load_bfrag_glb_f32(W2, col, kk * 32, half);

    const float thr1 = thr1p[0], leak1 = leak1p[0];
    const float thr2 = thr2p[0], leak2 = leak2p[0];
    const float it1 = 1.0f / thr1, it2 = 1.0f / thr2;
    const float b1v = b1[col], b2v = b2[col], b3v = b3[col];

    v8f m1[2] = {}, m2[2] = {}, acc[2] = {};   // persistent state, both row tiles

    const int xrow  = tid >> 4;                // 0..31 batch row (gather)
    const int xcol0 = (tid & 15) * 16;         // 16-wide column chunk
    __syncthreads();                           // sW3 ready

    for (int word = 0; word < SS; ++word) {
        // ---- gather embedding row (padding_idx 0 -> zeros), f32 -> bf16 ----
        const int tok = text[word * BB + xrow];
        const float* er = emb + (size_t)tok * HD;
        v16f xv = *(const v16f*)(er + xcol0);  // one 64B load
#pragma unroll
        for (int j = 0; j < 16; ++j) {
            float v = (tok == 0) ? 0.0f : xv[j];
            sX[xrow * HD + xcol0 + j] = (__bf16)v;
        }
        __syncthreads();

        // ---- delta1 = x @ W1^T + b1 (invariant over T, computed once) ----
        v8f d1[2] = {};
#pragma unroll
        for (int kk = 0; kk < 8; ++kk) {
            v16bf b = load_bfrag_glb_bf(W1bf, col, kk * 32, half);  // 32B, L2-hot
#pragma unroll
            for (int rt = 0; rt < 2; ++rt) {
                v16bf a = load_afrag(sX, rt * 16 + laneM, kk * 32, half);
                d1[rt] = __builtin_amdgcn_wmma_f32_16x16x32_bf16(
                    false, a, false, b, (short)0, d1[rt], false, false);
            }
        }
#pragma unroll
        for (int rt = 0; rt < 2; ++rt)
#pragma unroll
            for (int i = 0; i < 8; ++i) d1[rt][i] += b1v;

        // ---- T LIF timesteps ----
        for (int t = 0; t < TT; ++t) {
            // LIF layer 1 (soft reset) -> s1 spikes (bf16 bit pattern {0,1})
#pragma unroll
            for (int rt = 0; rt < 2; ++rt)
#pragma unroll
                for (int i = 0; i < 8; ++i) {
                    float m = leak1 * m1[rt][i] + d1[rt][i];
                    bool fire = (m * it1 - 1.0f) > 0.0f;
                    m1[rt][i] = fire ? (m - thr1) : m;
                    sS1u[(rt * 16 + i + 8 * half) * HD + col] =
                        fire ? ONE_BF : (unsigned short)0;
                }
            __syncthreads();

            // layer 2: s1 @ W2^T + b2 -> LIF2 -> s2 (W2 from registers)
            v8f c2[2] = {};
#pragma unroll
            for (int kk = 0; kk < 8; ++kk)
#pragma unroll
                for (int rt = 0; rt < 2; ++rt) {
                    v16bf a = load_afrag(sS1, rt * 16 + laneM, kk * 32, half);
                    c2[rt] = __builtin_amdgcn_wmma_f32_16x16x32_bf16(
                        false, a, false, Bw2[kk], (short)0, c2[rt], false, false);
                }
#pragma unroll
            for (int rt = 0; rt < 2; ++rt)
#pragma unroll
                for (int i = 0; i < 8; ++i) {
                    float m = leak2 * m2[rt][i] + c2[rt][i] + b2v;
                    bool fire = (m * it2 - 1.0f) > 0.0f;
                    m2[rt][i] = fire ? (m - thr2) : m;
                    sS2u[(rt * 16 + i + 8 * half) * HD + col] =
                        fire ? ONE_BF : (unsigned short)0;
                }
            __syncthreads();

            // integrator: acc += s2 @ W3^T + b3 (W3 from LDS, non-hoistable)
#pragma unroll
            for (int kk = 0; kk < 8; ++kk) {
                v16bf b = load_bfrag_lds(sW3, col, kk * 32, half);
#pragma unroll
                for (int rt = 0; rt < 2; ++rt) {
                    v16bf a = load_afrag(sS2, rt * 16 + laneM, kk * 32, half);
                    acc[rt] = __builtin_amdgcn_wmma_f32_16x16x32_bf16(
                        false, a, false, b, (short)0, acc[rt], false, false);
                }
            }
#pragma unroll
            for (int rt = 0; rt < 2; ++rt)
#pragma unroll
                for (int i = 0; i < 8; ++i) acc[rt][i] += b3v;
        }
    }

    // ---- ANN head: h = relu((acc/B) @ Wa1^T + ba1) ; out = h @ Wa2^T + ba2 ----
    const float invB = 1.0f / (float)BB;
#pragma unroll
    for (int rt = 0; rt < 2; ++rt)
#pragma unroll
        for (int i = 0; i < 8; ++i)
            sS1[(rt * 16 + i + 8 * half) * HD + col] = (__bf16)(acc[rt][i] * invB);
    __syncthreads();

    v8f hc[2] = {};
#pragma unroll
    for (int kk = 0; kk < 8; ++kk) {
        v16bf b = load_bfrag_glb_f32(Wa1, col, kk * 32, half);
#pragma unroll
        for (int rt = 0; rt < 2; ++rt) {
            v16bf a = load_afrag(sS1, rt * 16 + laneM, kk * 32, half);
            hc[rt] = __builtin_amdgcn_wmma_f32_16x16x32_bf16(
                false, a, false, b, (short)0, hc[rt], false, false);
        }
    }
    __syncthreads();                 // all waves done reading sS1 before hF overlays it

    float* hF = (float*)smem;        // 32 KB f32 h over sX+sS1 region
    const float ba1v = ba1[col];
#pragma unroll
    for (int rt = 0; rt < 2; ++rt)
#pragma unroll
        for (int i = 0; i < 8; ++i) {
            float h = hc[rt][i] + ba1v;
            hF[(rt * 16 + i + 8 * half) * HD + col] = (h > 0.0f) ? h : 0.0f;
        }
    __syncthreads();

    if (tid < BB) {
        float s = ba2[0];
        for (int j = 0; j < HD; ++j) s += hF[tid * HD + j] * Wa2[j];
        out[tid] = s;
    }
}

extern "C" void kernel_launch(void* const* d_in, const int* in_sizes, int n_in,
                              void* d_out, int out_size, void* d_ws, size_t ws_size,
                              hipStream_t stream) {
    const int*   text  = (const int*)  d_in[0];
    // d_in[1] = text_lengths (unused by the reference forward pass)
    const float* emb   = (const float*)d_in[2];
    const float* W1    = (const float*)d_in[3];
    const float* b1    = (const float*)d_in[4];
    const float* thr1  = (const float*)d_in[5];
    const float* leak1 = (const float*)d_in[6];
    const float* W2    = (const float*)d_in[7];
    const float* b2    = (const float*)d_in[8];
    const float* thr2  = (const float*)d_in[9];
    const float* leak2 = (const float*)d_in[10];
    const float* W3    = (const float*)d_in[11];
    const float* b3    = (const float*)d_in[12];
    const float* Wa1   = (const float*)d_in[13];
    const float* ba1   = (const float*)d_in[14];
    const float* Wa2   = (const float*)d_in[15];
    const float* ba2   = (const float*)d_in[16];
    float* out = (float*)d_out;

    // Stage W1 as bf16 in workspace (deterministic, recomputed every call).
    __bf16* W1bf = (__bf16*)d_ws;   // 128 KB
    W1_to_bf16_kernel<<<dim3(256), dim3(256), 0, stream>>>(W1, W1bf);

    const size_t shmem = 180224;  // 176 KB of the WGP's 320 KB LDS
    (void)hipFuncSetAttribute((const void*)SNN_persistent_kernel,
                              hipFuncAttributeMaxDynamicSharedMemorySize, (int)shmem);

    SNN_persistent_kernel<<<dim3(1), dim3(512), shmem, stream>>>(
        text, emb, W1bf, b1, thr1, leak1, W2, b2, thr2, leak2,
        W3, b3, Wa1, ba1, Wa2, ba2, out);
}
